// MultiHeadAttention_6219112644790
// MI455X (gfx1250) — compile-verified, matrix operand order checked by reading, not run on hardware
//
#include <hip/hip_runtime.h>
#include <hip/hip_bf16.h>

typedef __attribute__((ext_vector_type(16))) _Float16 v16h;
typedef __attribute__((ext_vector_type(8)))  _Float16 v8h;
typedef __attribute__((ext_vector_type(8)))  float    v8f;
typedef __attribute__((ext_vector_type(4)))  float    v4f;

#define WMMA16(a, b, c) \
  __builtin_amdgcn_wmma_f32_16x16x32_f16(false, (a), false, (b), (short)0, (c), false, false)

static constexpr int B_   = 8;
static constexpr int S_   = 1024;   // N + G
static constexpr int N_   = 1023;
static constexpr int HID_ = 512;
static constexpr int H_   = 8;
static constexpr int D_   = 64;

static __device__ inline v8f v8f_zero() {
  v8f z = {0.f, 0.f, 0.f, 0.f, 0.f, 0.f, 0.f, 0.f};
  return z;
}

static __device__ inline v16h pack16(v8h lo, v8h hi) {
  v16h r;
#pragma unroll
  for (int i = 0; i < 8; ++i) { r[i] = lo[i]; r[i + 8] = hi[i]; }
  return r;
}

// ---------------------------------------------------------------------------
// f32 -> f16 conversion (weights)
// ---------------------------------------------------------------------------
__global__ void cvt_f32_f16_kernel(const float* __restrict__ in,
                                   _Float16* __restrict__ out, int n) {
  int i = blockIdx.x * blockDim.x + threadIdx.x;
  if (i < n) out[i] = (_Float16)in[i];
}

// ---------------------------------------------------------------------------
// Projection: out = (X @ W^T + bias) * alpha, X:[B*S,512] f32, W16:[512,512] f16
// Output f16, either [b][h][s][d] (transposed=0) or [b][h][d][s] (transposed=1).
// One wave computes a 16(M) x 64(N) strip via 4 accumulators, K = 512.
// ---------------------------------------------------------------------------
__global__ void __launch_bounds__(128)
proj_qkv_kernel(const float* __restrict__ X,
                const _Float16* __restrict__ W16,
                const float* __restrict__ bias,
                _Float16* __restrict__ out,
                float alpha, int transposed) {
  const int lane  = threadIdx.x & 31;
  const int wave  = threadIdx.x >> 5;
  const int tile  = blockIdx.x * 4 + wave;     // 4096 tiles total
  const int mtile = tile >> 3;                 // 0..511  (B*S/16 rows)
  const int nstr  = tile & 7;                  // 0..7    (512/64 col strips)
  const int m0    = mtile * 16;
  const int n0    = nstr * 64;
  const int l16   = lane & 15;
  const int half8 = (lane >> 4) * 8;

  const float* xrow = X + (size_t)(m0 + l16) * HID_;

  v8f acc[4];
#pragma unroll
  for (int i = 0; i < 4; ++i) acc[i] = v8f_zero();

  for (int k0 = 0; k0 < HID_; k0 += 32) {
    const int ka = k0 + half8;
    const int kb = k0 + 16 + half8;
    // A fragment: X row, f32 -> f16 inline
    v4f xa0 = *(const v4f*)(xrow + ka);
    v4f xa1 = *(const v4f*)(xrow + ka + 4);
    v4f xb0 = *(const v4f*)(xrow + kb);
    v4f xb1 = *(const v4f*)(xrow + kb + 4);
    v16h a;
#pragma unroll
    for (int i = 0; i < 4; ++i) {
      a[i]      = (_Float16)xa0[i];
      a[i + 4]  = (_Float16)xa1[i];
      a[i + 8]  = (_Float16)xb0[i];
      a[i + 12] = (_Float16)xb1[i];
    }
#pragma unroll
    for (int nt = 0; nt < 4; ++nt) {
      const _Float16* wrow = W16 + (size_t)(n0 + nt * 16 + l16) * HID_;
      v8h b0 = *(const v8h*)(wrow + ka);
      v8h b1 = *(const v8h*)(wrow + kb);
      acc[nt] = WMMA16(a, pack16(b0, b1), acc[nt]);
    }
  }

#pragma unroll
  for (int nt = 0; nt < 4; ++nt) {
    const int j  = n0 + nt * 16 + l16;   // output column (h,d)
    const int h  = j >> 6;
    const int d  = j & 63;
    const float bj = bias[j];
#pragma unroll
    for (int r = 0; r < 8; ++r) {
      const int sg = m0 + half8 + r;     // global row b*S + s
      const int b  = sg >> 10;
      const int s  = sg & (S_ - 1);
      const float v = (acc[nt][r] + bj) * alpha;
      size_t idx;
      if (transposed)
        idx = ((size_t)((b * H_ + h) * D_ + d) << 10) + s;       // [b][h][d][s]
      else
        idx = (((size_t)(b * H_ + h) << 10) + s) * D_ + d;        // [b][h][s][d]
      out[idx] = (_Float16)v;
    }
  }
}

// ---------------------------------------------------------------------------
// Fused flash attention. Block = (b, qtile of 16), 8 waves = 8 heads.
// Computes S^T = K.Q^T (so lane = query), online softmax, O^T = V^T.P^T.
// ---------------------------------------------------------------------------
__global__ void __launch_bounds__(256)
attn_kernel(const _Float16* __restrict__ qh,   // [b][h][s][d], pre-scaled
            const _Float16* __restrict__ kh,   // [b][h][s][d]
            const _Float16* __restrict__ vt,   // [b][h][d][s]
            const int* __restrict__ attn_bias, // [b][N][N]
            const float* __restrict__ bias_table, // [256][8]
            const float* __restrict__ vbias,      // [1][8]
            _Float16* __restrict__ ctx) {         // [b][s][h*64+d]
  __shared__ float tbl[256 * H_];

  const int lane  = threadIdx.x & 31;
  const int h     = threadIdx.x >> 5;
  const int b     = blockIdx.x >> 6;
  const int qt    = blockIdx.x & 63;
  const int l16   = lane & 15;
  const int half8 = (lane >> 4) * 8;

  for (int i = threadIdx.x; i < 256 * H_; i += 256) tbl[i] = bias_table[i];
  __syncthreads();

  const float vb = vbias[h];
  const int   q  = qt * 16 + l16;           // query owned by this lane
  const size_t bh = (size_t)(b * H_ + h);

  // Q as B-fragments (lane n = query, K = d)
  const _Float16* qrow = qh + (bh * S_ + q) * D_;
  const v16h bq0 = pack16(*(const v8h*)(qrow + half8),
                          *(const v8h*)(qrow + 16 + half8));
  const v16h bq1 = pack16(*(const v8h*)(qrow + 32 + half8),
                          *(const v8h*)(qrow + 48 + half8));

  v8f o[4];
#pragma unroll
  for (int i = 0; i < 4; ++i) o[i] = v8f_zero();
  float m_run = -1.0e30f;
  float l_run = 0.f;

  const int* abrow = attn_bias + ((size_t)b * N_ + (q < N_ ? q : 0)) * N_;

  for (int t0 = 0; t0 < S_; t0 += 32) {
    if (q < N_ && t0 + 32 < N_)
      __builtin_prefetch(abrow + t0 + 32, 0, 1);   // global_prefetch_b8

    // ---- scores^T: two 16x16 tiles over this 32-key block ----
    v8f s[2];
#pragma unroll
    for (int tt = 0; tt < 2; ++tt) {
      s[tt] = v8f_zero();
      const int trow = t0 + tt * 16 + l16;
      const _Float16* kr = kh + (bh * S_ + trow) * D_;
      v16h a0 = pack16(*(const v8h*)(kr + half8),
                       *(const v8h*)(kr + 16 + half8));
      v16h a1 = pack16(*(const v8h*)(kr + 32 + half8),
                       *(const v8h*)(kr + 48 + half8));
      s[tt] = WMMA16(a0, bq0, s[tt]);
      s[tt] = WMMA16(a1, bq1, s[tt]);
    }

    // ---- structural bias + running max ----
    float mc = -1.0e30f;
#pragma unroll
    for (int tt = 0; tt < 2; ++tt) {
#pragma unroll
      for (int r = 0; r < 8; ++r) {
        const int t = t0 + tt * 16 + half8 + r;
        float bias;
        if (q < N_ && t < N_) {
          const int idx = abrow[t];
          bias = (idx == 255) ? -1.0e30f : tbl[idx * H_ + h];
        } else {
          bias = vb;   // global node row/column
        }
        const float sv = s[tt][r] + bias;
        s[tt][r] = sv;
        mc = fmaxf(mc, sv);
      }
    }
    mc = fmaxf(mc, __shfl_xor(mc, 16, 32));
    const float mnew  = fmaxf(m_run, mc);
    const float scal  = __expf(m_run - mnew);

    // ---- exponentiate; pack P^T directly as next B-fragment (lane-local) ----
    float lsum = 0.f;
    v16h pb;
#pragma unroll
    for (int tt = 0; tt < 2; ++tt) {
#pragma unroll
      for (int r = 0; r < 8; ++r) {
        const float p = __expf(s[tt][r] - mnew);
        lsum += p;
        pb[tt * 8 + r] = (_Float16)p;
      }
    }
    lsum += __shfl_xor(lsum, 16, 32);
    l_run = l_run * scal + lsum;
    m_run = mnew;

#pragma unroll
    for (int dt = 0; dt < 4; ++dt)
#pragma unroll
      for (int r = 0; r < 8; ++r) o[dt][r] *= scal;

    // ---- O^T += V^T . P^T ----
#pragma unroll
    for (int dt = 0; dt < 4; ++dt) {
      const int d = dt * 16 + l16;
      const _Float16* vr = vt + (bh * D_ + d) * S_ + t0;
      v16h av = pack16(*(const v8h*)(vr + half8),
                       *(const v8h*)(vr + 16 + half8));
      o[dt] = WMMA16(av, pb, o[dt]);
    }
  }

  // ---- normalize, write ctx [b][s][h*64+d] (16B per store) ----
  const float inv = 1.0f / l_run;
  const size_t cbase = ((size_t)(b * S_ + q)) * HID_ + h * D_;
#pragma unroll
  for (int dt = 0; dt < 4; ++dt) {
    v8h pkt;
#pragma unroll
    for (int r = 0; r < 8; ++r) pkt[r] = (_Float16)(o[dt][r] * inv);
    *(v8h*)(ctx + cbase + dt * 16 + half8) = pkt;
  }
}

// ---------------------------------------------------------------------------
// Output projection: out = ctx @ Wo^T + bo, f32 output [B*S, 512]
// ---------------------------------------------------------------------------
__global__ void __launch_bounds__(128)
proj_out_kernel(const _Float16* __restrict__ ctx,
                const _Float16* __restrict__ Wo16,
                const float* __restrict__ bo,
                float* __restrict__ out) {
  const int lane  = threadIdx.x & 31;
  const int wave  = threadIdx.x >> 5;
  const int tile  = blockIdx.x * 4 + wave;
  const int mtile = tile >> 3;
  const int nstr  = tile & 7;
  const int m0    = mtile * 16;
  const int n0    = nstr * 64;
  const int l16   = lane & 15;
  const int half8 = (lane >> 4) * 8;

  const _Float16* xrow = ctx + (size_t)(m0 + l16) * HID_;

  v8f acc[4];
#pragma unroll
  for (int i = 0; i < 4; ++i) acc[i] = v8f_zero();

  for (int k0 = 0; k0 < HID_; k0 += 32) {
    const int ka = k0 + half8;
    const int kb = k0 + 16 + half8;
    v16h a = pack16(*(const v8h*)(xrow + ka), *(const v8h*)(xrow + kb));
#pragma unroll
    for (int nt = 0; nt < 4; ++nt) {
      const _Float16* wrow = Wo16 + (size_t)(n0 + nt * 16 + l16) * HID_;
      v8h b0 = *(const v8h*)(wrow + ka);
      v8h b1 = *(const v8h*)(wrow + kb);
      acc[nt] = WMMA16(a, pack16(b0, b1), acc[nt]);
    }
  }

#pragma unroll
  for (int nt = 0; nt < 4; ++nt) {
    const int j = n0 + nt * 16 + l16;
    const float bj = bo[j];
#pragma unroll
    for (int r = 0; r < 8; ++r) {
      const int sg = m0 + half8 + r;
      out[(size_t)sg * HID_ + j] = acc[nt][r] + bj;
    }
  }
}

// ---------------------------------------------------------------------------
extern "C" void kernel_launch(void* const* d_in, const int* in_sizes, int n_in,
                              void* d_out, int out_size, void* d_ws, size_t ws_size,
                              hipStream_t stream) {
  const float* q   = (const float*)d_in[0];
  const float* k   = (const float*)d_in[1];
  const float* v   = (const float*)d_in[2];
  const int*   ab  = (const int*)d_in[3];
  const float* Wq  = (const float*)d_in[4];
  const float* bq  = (const float*)d_in[5];
  const float* Wk  = (const float*)d_in[6];
  const float* bk  = (const float*)d_in[7];
  const float* Wv  = (const float*)d_in[8];
  const float* bv  = (const float*)d_in[9];
  const float* Wo  = (const float*)d_in[10];
  const float* bo  = (const float*)d_in[11];
  const float* bt  = (const float*)d_in[12];
  const float* vbs = (const float*)d_in[13];
  float* out = (float*)d_out;

  // Workspace layout (f16 intermediates; ~34 MB, fits in L2)
  char* ws = (char*)d_ws;
  const size_t WSZ  = (size_t)HID_ * HID_ * sizeof(_Float16);   // 512 KB
  const size_t HSZ  = (size_t)B_ * H_ * S_ * D_ * sizeof(_Float16); // 8 MB
  _Float16* Wq16 = (_Float16*)(ws);
  _Float16* Wk16 = (_Float16*)(ws + WSZ);
  _Float16* Wv16 = (_Float16*)(ws + 2 * WSZ);
  _Float16* Wo16 = (_Float16*)(ws + 3 * WSZ);
  _Float16* qhp  = (_Float16*)(ws + 4 * WSZ);
  _Float16* khp  = (_Float16*)(ws + 4 * WSZ + HSZ);
  _Float16* vtp  = (_Float16*)(ws + 4 * WSZ + 2 * HSZ);
  _Float16* ctx  = (_Float16*)(ws + 4 * WSZ + 3 * HSZ);

  const int nW = HID_ * HID_;
  cvt_f32_f16_kernel<<<(nW + 255) / 256, 256, 0, stream>>>(Wq, Wq16, nW);
  cvt_f32_f16_kernel<<<(nW + 255) / 256, 256, 0, stream>>>(Wk, Wk16, nW);
  cvt_f32_f16_kernel<<<(nW + 255) / 256, 256, 0, stream>>>(Wv, Wv16, nW);
  cvt_f32_f16_kernel<<<(nW + 255) / 256, 256, 0, stream>>>(Wo, Wo16, nW);

  const float scale = 0.125f;  // 1/sqrt(64)
  // 4096 (16x64) tiles, 4 waves/block -> 1024 blocks
  proj_qkv_kernel<<<1024, 128, 0, stream>>>(q, Wq16, bq, qhp, scale, 0);
  proj_qkv_kernel<<<1024, 128, 0, stream>>>(k, Wk16, bk, khp, 1.0f, 0);
  proj_qkv_kernel<<<1024, 128, 0, stream>>>(v, Wv16, bv, vtp, 1.0f, 1);

  // 8 batches x 64 query tiles; 8 heads = 8 waves per block
  attn_kernel<<<B_ * 64, 256, 0, stream>>>(qhp, khp, vtp, ab, bt, vbs, ctx);

  proj_out_kernel<<<1024, 128, 0, stream>>>(ctx, Wo16, bo, out);
}
